// DanceDecoder_7636451852328
// MI455X (gfx1250) — matrix operand order repaired and checked by reading, not compiled, not run on hardware
//
#include <hip/hip_runtime.h>
#include <hip/hip_bf16.h>

#define DEVINL __device__ __forceinline__

typedef __attribute__((ext_vector_type(16))) __bf16 v16bf;
typedef __attribute__((ext_vector_type(8)))  float  v8f;
typedef unsigned short u16;

union FragU { v16bf v; uint4 q[2]; };
union VecU  { uint4 q; u16 s[8]; };

DEVINL u16 f2bf(float x) {
  unsigned int u = __float_as_uint(x);
  unsigned int r = u + 0x7FFFu + ((u >> 16) & 1u);
  return (u16)(r >> 16);
}
DEVINL float bf2f(u16 h) { return __uint_as_float((unsigned int)h << 16); }
DEVINL unsigned int pack2bf(float lo, float hi) {
  return (unsigned int)f2bf(lo) | ((unsigned int)f2bf(hi) << 16);
}

DEVINL v8f wmma_bf16(v16bf a, v16bf b, v8f c) {
  return __builtin_amdgcn_wmma_f32_16x16x32_bf16(false, a, false, b, (short)0, c, false, false);
}

// ---- CDNA5 async global->LDS copy (16B per lane), ASYNCcnt tracked ----
DEVINL void async_copy16(void* lds_ptr, const void* gptr) {
  unsigned lds_off = (unsigned)(size_t)lds_ptr;          // low 32b of flat LDS addr
  unsigned long long ga = (unsigned long long)(size_t)gptr;
  asm volatile("global_load_async_to_lds_b128 %0, %1, off"
               :: "v"(lds_off), "v"(ga) : "memory");
}
DEVINL void async_wait0() {
  asm volatile("s_wait_asynccnt 0x0" ::: "memory");
}

// A fragment: 16(M) x 32(K) bf16.  lane<16: K 0-15 ; lane>=16: K 16-31
DEVINL v16bf frag_a(const u16* base, int pitch, int rowbase, int kofs, int lane) {
  int m  = rowbase + (lane & 15);
  int ko = (lane < 16) ? 0 : 16;
  const u16* p = base + m * pitch + kofs;
  FragU f;
  f.q[0] = *(const uint4*)(p + ko);
  f.q[1] = *(const uint4*)(p + 8 + ko);
  return f.v;
}
// B fragment: 32(K) x 16(N) bf16, LDS holds B^T as Bs[n][k].
DEVINL v16bf frag_b(const u16* base, int pitch, int nbase, int kofs, int lane) {
  int n  = nbase + (lane & 15);
  int ko = (lane < 16) ? 0 : 16;
  const u16* p = base + n * pitch + kofs + ko;
  FragU f;
  f.q[0] = *(const uint4*)(p);
  f.q[1] = *(const uint4*)(p + 8);
  return f.v;
}

// ---------------------------------------------------------------------------
// GEMM: C = act( A[MxK](bf16) @ W^T[KxN](bf16, row-major (N,K)) + bias(f32) )
// Double-buffered LDS; async DMA staging overlapped with WMMA.
// ---------------------------------------------------------------------------
#define FLAG_RELU   2
#define FLAG_ADD    4
#define FLAG_BF16O  8

#define BM 128
#define BN 64
#define BK 64
#define LP 72   // LDS pitch (halves): 144 bytes

DEVINL void gemm_load_tiles(u16* Asb, u16* Bsb,
                            const u16* A, int lda, int Mtot, int m0,
                            const u16* W, int ldw, int Ntot, int n0,
                            int Ktot, int k0, bool mfull, bool nfull, int tid)
{
  const bool kfull = (k0 + BK) <= Ktot;
  if (mfull && kfull) {
    #pragma unroll
    for (int p = 0; p < 4; ++p) {
      int idx = tid + p * 256;
      int r = idx >> 3, c8 = (idx & 7) * 8;
      async_copy16(&Asb[r * LP + c8], A + (size_t)(m0 + r) * lda + k0 + c8);
    }
  } else {
    for (int i = tid; i < BM * BK; i += 256) {
      int r = i >> 6, k = i & 63;
      int gm = m0 + r, gk = k0 + k;
      Asb[r * LP + k] = (gm < Mtot && gk < Ktot) ? A[(size_t)gm * lda + gk] : (u16)0;
    }
  }
  if (nfull && kfull) {
    #pragma unroll
    for (int p = 0; p < 2; ++p) {
      int idx = tid + p * 256;
      int r = idx >> 3, c8 = (idx & 7) * 8;
      async_copy16(&Bsb[r * LP + c8], W + (size_t)(n0 + r) * ldw + k0 + c8);
    }
  } else {
    for (int i = tid; i < BN * BK; i += 256) {
      int n = i >> 6, k = i & 63;
      int gn = n0 + n, gk = k0 + k;
      Bsb[n * LP + k] = (gn < Ntot && gk < Ktot) ? W[(size_t)gn * ldw + gk] : (u16)0;
    }
  }
}

__global__ __launch_bounds__(256) void gemm_bf16_kernel(
    const u16* __restrict__ A, int lda, int Mtot,
    const u16* __restrict__ W, int ldw, int Ntot, int Ktot,
    const float* __restrict__ bias,
    void* __restrict__ Cv, int ldc, int flags)
{
  __shared__ __align__(16) u16 As[2][BM * LP];
  __shared__ __align__(16) u16 Bs[2][BN * LP];

  const int tid  = threadIdx.x;
  const int lane = tid & 31;
  const int wave = tid >> 5;
  const int m0 = blockIdx.y * BM;
  const int n0 = blockIdx.x * BN;
  const int wm = (wave & 3) * 32;
  const int wn = (wave >> 2) * 32;

  const bool mfull = (m0 + BM) <= Mtot;
  const bool nfull = (n0 + BN) <= Ntot;

  v8f c00 = {}, c01 = {}, c10 = {}, c11 = {};

  const int nk = (Ktot + BK - 1) / BK;
  gemm_load_tiles(As[0], Bs[0], A, lda, Mtot, m0, W, ldw, Ntot, n0, Ktot, 0,
                  mfull, nfull, tid);

  for (int kb = 0; kb < nk; ++kb) {
    const int cur = kb & 1;
    async_wait0();
    __syncthreads();
    if (kb + 1 < nk)
      gemm_load_tiles(As[cur ^ 1], Bs[cur ^ 1], A, lda, Mtot, m0,
                      W, ldw, Ntot, n0, Ktot, (kb + 1) * BK, mfull, nfull, tid);

    const u16* Ac = As[cur];
    const u16* Bc = Bs[cur];
    #pragma unroll
    for (int ks = 0; ks < 2; ++ks) {
      int kofs = ks * 32;
      v16bf a0 = frag_a(Ac, LP, wm,      kofs, lane);
      v16bf a1 = frag_a(Ac, LP, wm + 16, kofs, lane);
      v16bf b0 = frag_b(Bc, LP, wn,      kofs, lane);
      v16bf b1 = frag_b(Bc, LP, wn + 16, kofs, lane);
      c00 = wmma_bf16(a0, b0, c00);
      c01 = wmma_bf16(a0, b1, c01);
      c10 = wmma_bf16(a1, b0, c10);
      c11 = wmma_bf16(a1, b1, c11);
    }
  }

  float* Cf = (float*)Cv;
  u16*   Ch = (u16*)Cv;
  v8f acc[2][2] = {{c00, c01}, {c10, c11}};
  #pragma unroll
  for (int it = 0; it < 2; ++it) {
    #pragma unroll
    for (int jt = 0; jt < 2; ++jt) {
      int ncol = n0 + wn + jt * 16 + (lane & 15);
      bool nok = ncol < Ntot;
      float bv = (bias && nok) ? bias[ncol] : 0.f;
      #pragma unroll
      for (int r = 0; r < 8; ++r) {
        int mrow = m0 + wm + it * 16 + ((lane < 16) ? r : r + 8);
        if ((mfull || mrow < Mtot) && (nfull || nok)) {
          float v = acc[it][jt][r] + bv;
          if (flags & FLAG_RELU) v = fmaxf(v, 0.f);
          size_t idx = (size_t)mrow * ldc + ncol;
          if (flags & FLAG_BF16O)      Ch[idx] = f2bf(v);
          else if (flags & FLAG_ADD)   Cf[idx] += v;
          else                         Cf[idx] = v;
        }
      }
    }
  }
}

// ---------------------------------------------------------------------------
// Flash-style attention, all-bf16 I/O, dh = 64, strides in halves.
// ---------------------------------------------------------------------------
__global__ __launch_bounds__(256) void attn_kernel(
    const u16* __restrict__ qb, int qld, int qoff,
    const u16* __restrict__ kb, int kld, int koff,
    const u16* __restrict__ vb, int vld, int voff,
    u16* __restrict__ ob, int oldc,
    int Nq, int Nk, int H, float scale)
{
  const int bh = blockIdx.y;
  const int b = bh / H, h = bh % H;
  const int q0 = blockIdx.x * 64;

  __shared__ __align__(16) u16 Qs[64 * 72];
  __shared__ __align__(16) u16 Ks[32 * 72];
  __shared__ __align__(16) u16 Vts[64 * 40];  // V^T: [d][key]
  __shared__ __align__(16) u16 Ps[64 * 40];
  __shared__ float Ss[64 * 32];
  __shared__ float rmax[64], rsum[64], ralpha[64];

  const int tid = threadIdx.x, lane = tid & 31, wave = tid >> 5;
  const int wrow = (wave & 3) * 16;
  const int wcol = (wave >> 2) * 32;

  #pragma unroll
  for (int p = 0; p < 2; ++p) {        // Q: 64 rows x 8 b128 async copies
    int idx = tid + p * 256;
    int r = idx >> 3, c8 = (idx & 7) * 8;
    int gq = q0 + r; if (gq >= Nq) gq = Nq - 1;   // clamp; OOB rows dropped at store
    async_copy16(&Qs[r * 72 + c8], qb + (size_t)(b * Nq + gq) * qld + qoff + h * 64 + c8);
  }
  if (tid < 64) { rmax[tid] = -3e38f; rsum[tid] = 0.f; }

  v8f o0 = {}, o1 = {};
  const int nkb = (Nk + 31) / 32;
  for (int kbk = 0; kbk < nkb; ++kbk) {
    const int kk0 = kbk * 32;
    {
      int r = tid >> 3, c8 = (tid & 7) * 8;
      int gk = kk0 + r; if (gk >= Nk) gk = Nk - 1;   // clamp; masked in S / zero-weighted in P
      async_copy16(&Ks[r * 72 + c8], kb + (size_t)(b * Nk + gk) * kld + koff + h * 64 + c8);
      VecU vu;
      vu.q = *(const uint4*)(vb + (size_t)(b * Nk + gk) * vld + voff + h * 64 + c8);
      #pragma unroll
      for (int j = 0; j < 8; ++j) Vts[(c8 + j) * 40 + r] = vu.s[j];
    }
    async_wait0();
    __syncthreads();

    {   // S tile 64x32: wave w -> rows 16*(w&3), keys 16*(w>>2)
      int sr = (wave & 3) * 16, sc = (wave >> 2) * 16;
      v8f s = {};
      s = wmma_bf16(frag_a(Qs, 72, sr, 0, lane),  frag_b(Ks, 72, sc, 0, lane),  s);
      s = wmma_bf16(frag_a(Qs, 72, sr, 32, lane), frag_b(Ks, 72, sc, 32, lane), s);
      int keyl = sc + (lane & 15);
      bool oob = (kk0 + keyl) >= Nk;
      #pragma unroll
      for (int r = 0; r < 8; ++r) {
        int rr = sr + ((lane < 16) ? r : r + 8);
        Ss[rr * 32 + keyl] = oob ? -3e38f : s[r] * scale;
      }
    }
    __syncthreads();

    if (tid < 64) {
      float mb = -3e38f;
      for (int k = 0; k < 32; ++k) mb = fmaxf(mb, Ss[tid * 32 + k]);
      float nm = fmaxf(rmax[tid], mb);
      float al = __expf(rmax[tid] - nm);
      float s_ = 0.f;
      for (int k = 0; k < 32; k += 2) {
        float p0 = __expf(Ss[tid * 32 + k] - nm);
        float p1 = __expf(Ss[tid * 32 + k + 1] - nm);
        s_ += p0 + p1;
        *(unsigned int*)(&Ps[tid * 40 + k]) = pack2bf(p0, p1);
      }
      rsum[tid] = rsum[tid] * al + s_;
      rmax[tid] = nm;
      ralpha[tid] = al;
    }
    __syncthreads();

    #pragma unroll
    for (int r = 0; r < 8; ++r) {
      float al = ralpha[wrow + ((lane < 16) ? r : r + 8)];
      o0[r] *= al; o1[r] *= al;
    }
    v16bf ap = frag_a(Ps, 40, wrow, 0, lane);
    o0 = wmma_bf16(ap, frag_b(Vts, 40, wcol,      0, lane), o0);
    o1 = wmma_bf16(ap, frag_b(Vts, 40, wcol + 16, 0, lane), o1);
    __syncthreads();
  }

  #pragma unroll
  for (int r = 0; r < 8; ++r) {
    int mrl = wrow + ((lane < 16) ? r : r + 8);
    int gq = q0 + mrl;
    if (gq < Nq) {
      float inv = 1.f / rsum[mrl];
      size_t base = (size_t)(b * Nq + gq) * oldc + h * 64;
      ob[base + wcol + (lane & 15)]      = f2bf(o0[r] * inv);
      ob[base + wcol + 16 + (lane & 15)] = f2bf(o1[r] * inv);
    }
  }
}

// ---------------------------------------------------------------------------
// LayerNorm D=512, fp32 in -> bf16 out (packed pair stores)
// ---------------------------------------------------------------------------
__global__ __launch_bounds__(256) void ln_kernel(
    const float* __restrict__ x, const float* __restrict__ g,
    const float* __restrict__ bta, u16* __restrict__ y)
{
  __shared__ float red[256];
  __shared__ float stat[2];
  int row = blockIdx.x, tid = threadIdx.x;
  const float* xr = x + (size_t)row * 512;
  float v0 = xr[2 * tid], v1 = xr[2 * tid + 1];
  red[tid] = v0 + v1;
  __syncthreads();
  for (int o = 128; o > 0; o >>= 1) { if (tid < o) red[tid] += red[tid + o]; __syncthreads(); }
  if (tid == 0) stat[0] = red[0] * (1.f / 512.f);
  __syncthreads();
  float mean = stat[0];
  float d0 = v0 - mean, d1 = v1 - mean;
  red[tid] = d0 * d0 + d1 * d1;
  __syncthreads();
  for (int o = 128; o > 0; o >>= 1) { if (tid < o) red[tid] += red[tid + o]; __syncthreads(); }
  if (tid == 0) stat[1] = rsqrtf(red[0] * (1.f / 512.f) + 1e-5f);
  __syncthreads();
  float rstd = stat[1];
  float y0 = d0 * rstd * g[2 * tid]     + bta[2 * tid];
  float y1 = d1 * rstd * g[2 * tid + 1] + bta[2 * tid + 1];
  *(unsigned int*)(&y[(size_t)row * 512 + 2 * tid]) = pack2bf(y0, y1);
}

// ---------------------------------------------------------------------------
// Per-token 24x24 kinematic attention on bf16 buffers
// ---------------------------------------------------------------------------
__global__ __launch_bounds__(32) void ds_small_kernel(
    const u16* __restrict__ comb, u16* __restrict__ vbuf)
{
  int inst = blockIdx.x;
  int token = inst >> 2, h = inst & 3;
  const u16* qp = comb + (size_t)token * 1248 + h * 156 + 3;
  const u16* kp = qp + 624;
  u16* vp = vbuf + (size_t)token * 624 + h * 156 + 3;

  __shared__ float ks[24][6], vs[24][6], p[24][24];
  int t = threadIdx.x;
  if (t < 24) {
    #pragma unroll
    for (int j = 0; j < 6; ++j) { ks[t][j] = bf2f(kp[t * 6 + j]); vs[t][j] = bf2f(vp[t * 6 + j]); }
  }
  __syncthreads();
  if (t < 24) {
    float q[6];
    #pragma unroll
    for (int j = 0; j < 6; ++j) q[j] = bf2f(qp[t * 6 + j]);
    float s[24], mx = -3e38f;
    for (int k = 0; k < 24; ++k) {
      float d = 0.f;
      #pragma unroll
      for (int j = 0; j < 6; ++j) d += q[j] * ks[k][j];
      d *= 0.4082482904638631f;
      s[k] = d; mx = fmaxf(mx, d);
    }
    float sum = 0.f;
    for (int k = 0; k < 24; ++k) { float e = __expf(s[k] - mx); s[k] = e; sum += e; }
    float inv = 1.f / sum;
    for (int k = 0; k < 24; ++k) p[t][k] = s[k] * inv;
  }
  __syncthreads();
  if (t == 0) {
    #define U(i,j,si,sj) p[i][j] = (p[i][j] + p[si][sj]) * 0.5f
    U(0,6,0,3);  U(6,0,3,0);  U(0,9,0,6);  U(9,0,6,0);
    U(0,12,0,9); U(12,0,9,0); U(0,13,0,9); U(13,0,9,0); U(0,14,0,9); U(14,0,9,0);
    U(0,16,0,13);U(16,0,13,0);U(0,17,0,14);U(17,0,14,0);U(0,15,0,12);U(15,0,12,0);
    #undef U
  }
  __syncthreads();
  if (t < 24) {
    #pragma unroll
    for (int j = 0; j < 6; ++j) {
      float o = 0.f;
      for (int k = 0; k < 24; ++k) o += p[t][k] * vs[k][j];
      vp[t * 6 + j] = f2bf(o);
    }
  }
}

// ---------------------------------------------------------------------------
// conversion / elementwise helpers
// ---------------------------------------------------------------------------
__global__ __launch_bounds__(256) void cvt_kernel(const float* __restrict__ in,
                                                  u16* __restrict__ out, int n4)
{
  int i = blockIdx.x * 256 + threadIdx.x;
  if (i < n4) {
    const float4 v = *(const float4*)(in + 4 * (size_t)i);
    uint2 u; u.x = pack2bf(v.x, v.y); u.y = pack2bf(v.z, v.w);
    *(uint2*)(&out[4 * (size_t)i]) = u;
  }
}

__global__ __launch_bounds__(256) void tconv_kernel(const float* __restrict__ in,
                                                    u16* __restrict__ out, int R, int C)
{
  int i = blockIdx.x * 256 + threadIdx.x;
  if (i < R * C) {
    int r = i / C, c = i - r * C;
    out[(size_t)c * R + r] = f2bf(in[i]);
  }
}

__global__ __launch_bounds__(256) void pool_kernel(const float* __restrict__ x, float* __restrict__ pooled)
{
  int i = blockIdx.x * 256 + threadIdx.x;
  if (i < 128 * 512) {
    int b = i >> 9, d = i & 511;
    const float* p = x + (size_t)b * 300 * 512 + d;
    float s = 0.f;
    for (int n = 0; n < 300; ++n) s += p[(size_t)n * 512];
    pooled[i] = s * (1.f / 300.f);
  }
}

__global__ __launch_bounds__(256) void mish_kernel(const float* __restrict__ t,
                                                   const float* __restrict__ pooled,
                                                   u16* __restrict__ out)
{
  int i = blockIdx.x * 256 + threadIdx.x;
  if (i < 128 * 512) {
    float v = t[i] + pooled[i];
    float sp = (v > 20.f) ? v : log1pf(__expf(v));
    out[i] = f2bf(v * tanhf(sp));
  }
}

__global__ __launch_bounds__(256) void align_apply_kernel(float* __restrict__ out,
                                                          const float* __restrict__ xd,
                                                          const float* __restrict__ h)
{
  int i = blockIdx.x * 256 + threadIdx.x;
  if (i < 128 * 300 * 512) {
    int b = i / (300 * 512);
    int d = i & 511;
    float w  = h[b * 1024 + d];
    float bi = h[b * 1024 + 512 + d];
    out[i] += (w + 1.f) * xd[i] + bi;
  }
}

// ---------------------------------------------------------------------------
// host orchestration
// ---------------------------------------------------------------------------
static void run_gemm(const u16* A, int lda, int M, const u16* W, int ldw,
                     int N, int K, const float* bias, void* C, int ldc,
                     int flags, hipStream_t s)
{
  dim3 grid((N + BN - 1) / BN, (M + BM - 1) / BM);
  gemm_bf16_kernel<<<grid, 256, 0, s>>>(A, lda, M, W, ldw, N, K, bias, C, ldc, flags);
}

extern "C" void kernel_launch(void* const* d_in, const int* in_sizes, int n_in,
                              void* d_out, int out_size, void* d_ws, size_t ws_size,
                              hipStream_t stream)
{
  const float* tgt      = (const float*)d_in[0];
  const float* memory   = (const float*)d_in[1];
  const float* tvec     = (const float*)d_in[2];
  const float* qk_w     = (const float*)d_in[3];
  const float* v_w      = (const float*)d_in[4];
  const float* ds_lw    = (const float*)d_in[5];
  const float* ds_lb    = (const float*)d_in[6];
  const float* ta_in_w  = (const float*)d_in[7];
  const float* ta_in_b  = (const float*)d_in[8];
  const float* ta_out_w = (const float*)d_in[9];
  const float* ta_out_b = (const float*)d_in[10];
  const float* ca_in_w  = (const float*)d_in[11];
  const float* ca_in_b  = (const float*)d_in[12];
  const float* ca_out_w = (const float*)d_in[13];
  const float* ca_out_b = (const float*)d_in[14];
  const float* l1_w     = (const float*)d_in[15];
  const float* l1_b     = (const float*)d_in[16];
  const float* l2_w     = (const float*)d_in[17];
  const float* l2_b     = (const float*)d_in[18];
  const float* n1_g = (const float*)d_in[19]; const float* n1_b = (const float*)d_in[20];
  const float* n2_g = (const float*)d_in[21]; const float* n2_b = (const float*)d_in[22];
  const float* n3_g = (const float*)d_in[23]; const float* n3_b = (const float*)d_in[24];
  const float* n4_g = (const float*)d_in[25]; const float* n4_b = (const float*)d_in[26];
  const float* a1_w = (const float*)d_in[27]; const float* a1_b = (const float*)d_in[28];
  const float* a2_w = (const float*)d_in[29]; const float* a2_b = (const float*)d_in[30];
  const float* a3_w = (const float*)d_in[31]; const float* a3_b = (const float*)d_in[32];

  float* out = (float*)d_out;
  const int TOK = 128 * 300;
  const size_t DTOK = (size_t)TOK * 512;

  // fp32 region
  float* ws     = (float*)d_ws;
  float* xf     = ws;                                  // 38400*512  (x2/x3/x4)
  float* pooled = xf + DTOK;                           // 65536
  float* hbuf   = pooled + 128 * 512;                  // 131072
  // bf16 region
  u16* hp   = (u16*)(hbuf + 128 * 1024);
  u16* xnh  = hp;                  hp += DTOK;                     // ln out
  u16* bigh = hp;                  hp += (size_t)TOK * 2048;       // proj out
  u16* midh = hp;                  hp += (size_t)TOK * 640;        // v / attn out
  u16* memh = hp;                  hp += DTOK;                     // memory bf16
  u16* mishh = hp;                 hp += 128 * 512;
  u16* wqkT = hp;                  hp += (size_t)1248 * 512;
  u16* wvT  = hp;                  hp += (size_t)624 * 512;
  u16* dslwh = hp;                 hp += (size_t)512 * 624;
  u16* tainh = hp;                 hp += (size_t)1536 * 512;
  u16* taouth = hp;                hp += (size_t)512 * 512;
  u16* cainh = hp;                 hp += (size_t)1536 * 512;
  u16* caouth = hp;                hp += (size_t)512 * 512;
  u16* l1h  = hp;                  hp += (size_t)2048 * 512;
  u16* l2h  = hp;                  hp += (size_t)512 * 2048;
  u16* a1h  = hp;                  hp += (size_t)1024 * 512;
  u16* a2h  = hp;                  hp += (size_t)1024 * 512;
  u16* a3h  = hp;                  hp += (size_t)1024 * 512;
  u16* bigAh = bigh;
  u16* bigBh = bigh + DTOK;

  const int ELEM_BLK = (TOK * 512 + 255) / 256;
  auto cvt = [&](const float* src, u16* dst, size_t n) {
    cvt_kernel<<<(int)((n / 4 + 255) / 256), 256, 0, stream>>>(src, dst, (int)(n / 4));
  };

  auto align_block = [&](const float* xd, const u16* awh, const float* ab) {
    pool_kernel<<<(128 * 512 + 255) / 256, 256, 0, stream>>>(xd, pooled);
    mish_kernel<<<(128 * 512 + 255) / 256, 256, 0, stream>>>(tvec, pooled, mishh);
    run_gemm(mishh, 512, 128, awh, 512, 1024, 512, ab, hbuf, 1024, 0, stream);
    align_apply_kernel<<<ELEM_BLK, 256, 0, stream>>>(out, xd, hbuf);
  };

  // one-time conversions (weights + memory -> bf16; einsum weights transposed)
  tconv_kernel<<<(512 * 1248 + 255) / 256, 256, 0, stream>>>(qk_w, wqkT, 512, 1248);
  tconv_kernel<<<(512 * 624 + 255) / 256, 256, 0, stream>>>(v_w, wvT, 512, 624);
  cvt(ds_lw,    dslwh,  (size_t)512 * 624);
  cvt(ta_in_w,  tainh,  (size_t)1536 * 512);
  cvt(ta_out_w, taouth, (size_t)512 * 512);
  cvt(ca_in_w,  cainh,  (size_t)1536 * 512);
  cvt(ca_out_w, caouth, (size_t)512 * 512);
  cvt(l1_w,     l1h,    (size_t)2048 * 512);
  cvt(l2_w,     l2h,    (size_t)512 * 2048);
  cvt(a1_w,     a1h,    (size_t)1024 * 512);
  cvt(a2_w,     a2h,    (size_t)1024 * 512);
  cvt(a3_w,     a3h,    (size_t)1024 * 512);
  cvt(memory,   memh,   DTOK);

  // x = tgt (residual accumulator lives in d_out)
  hipMemcpyAsync(out, tgt, DTOK * sizeof(float), hipMemcpyDeviceToDevice, stream);

  // ---- phase 1: ds_attn ----
  ln_kernel<<<TOK, 256, 0, stream>>>(out, n1_g, n1_b, xnh);
  run_gemm(xnh, 512, TOK, wqkT, 512, 1248, 512, nullptr, bigh, 1248, FLAG_BF16O, stream);
  run_gemm(xnh, 512, TOK, wvT,  512,  624, 512, nullptr, midh,  624, FLAG_BF16O, stream);
  ds_small_kernel<<<TOK * 4, 32, 0, stream>>>(bigh, midh);
  run_gemm(midh, 624, TOK, dslwh, 624, 512, 624, ds_lb, out, 512, FLAG_ADD, stream);

  // ---- phase 2: self-attention + align(a1) ----
  ln_kernel<<<TOK, 256, 0, stream>>>(out, n2_g, n2_b, xnh);
  run_gemm(xnh, 512, TOK, tainh, 512, 1536, 512, ta_in_b, bigh, 1536, FLAG_BF16O, stream);
  attn_kernel<<<dim3(5, 128 * 8), 256, 0, stream>>>(
      bigh, 1536, 0, bigh, 1536, 512, bigh, 1536, 1024, midh, 512, 300, 300, 8, 0.125f);
  run_gemm(midh, 512, TOK, taouth, 512, 512, 512, ta_out_b, xf, 512, 0, stream);
  align_block(xf, a1h, a1_b);

  // ---- phase 3: cross-attention + align(a2) ----
  ln_kernel<<<TOK, 256, 0, stream>>>(out, n3_g, n3_b, xnh);
  run_gemm(xnh, 512, TOK, cainh, 512, 512, 512, ca_in_b, bigAh, 512, FLAG_BF16O, stream);
  run_gemm(memh, 512, TOK, cainh + (size_t)512 * 512, 512, 1024, 512,
           ca_in_b + 512, bigBh, 1024, FLAG_BF16O, stream);
  attn_kernel<<<dim3(5, 128 * 8), 256, 0, stream>>>(
      bigAh, 512, 0, bigBh, 1024, 0, bigBh, 1024, 512, midh, 512, 300, 300, 8, 0.125f);
  run_gemm(midh, 512, TOK, caouth, 512, 512, 512, ca_out_b, xf, 512, 0, stream);
  align_block(xf, a2h, a2_b);

  // ---- phase 4: FFN + align(a3) ----
  ln_kernel<<<TOK, 256, 0, stream>>>(out, n4_g, n4_b, xnh);
  run_gemm(xnh, 512, TOK, l1h, 512, 2048, 512, l1_b, bigh, 2048, FLAG_RELU | FLAG_BF16O, stream);
  run_gemm(bigh, 2048, TOK, l2h, 2048, 512, 2048, l2_b, xf, 512, 0, stream);
  align_block(xf, a3h, a3_b);
}
